// RelativeGlobalAttention_16286515987011
// MI455X (gfx1250) — matrix operand order repaired and checked by reading, not compiled, not run on hardware
//
#include <hip/hip_runtime.h>
#include <hip/hip_bf16.h>

// ---------------------------------------------------------------------------
// RelativeGlobalAttention for MI455X (gfx1250), wave32 + v_wmma_f32_16x16x32_f16
// Pipeline:
//   1) fp32 -> fp16 conversion of activations/weights/E into workspace
//   2) projection GEMMs: q (pre-scaled 1/sqrt(DH)), k -> head-major fp16;
//      v -> head-major TRANSPOSED fp16 (vT[bh][dh][l]) so the attn*V stage
//      reads its WMMA B-operand with contiguous 128-bit loads
//   3) fused score(QK^T + skewed Q.E^T) + causal mask + softmax + attn*V,
//      logits kept entirely in LDS (~129KB row panel), attn written once
//   4) fc GEMM -> fp32 output
// ---------------------------------------------------------------------------

typedef __attribute__((ext_vector_type(16))) _Float16 v16h;
typedef __attribute__((ext_vector_type(8)))  float    v8f;

static constexpr int B_ = 2, L_ = 2048, D_ = 512, H_ = 8, DH_ = 64;
static constexpr int SSTRIDE = L_ + 16;           // padded LDS row stride (floats)

__device__ __forceinline__ v8f wmma_f16(v16h a, v16h b, v8f c) {
  return __builtin_amdgcn_wmma_f32_16x16x32_f16(false, a, false, b, (short)0, c,
                                                false, false);
}

// A-matrix fragment (16x32 f16, ISA 7.12.2): row = lane&15, half = lane>>4,
// e<8 -> k = half*8+e ; e>=8 -> k = 16 + half*8 + (e-8)
__device__ __forceinline__ v16h frag_a_ld(const _Float16* p, int ld) {
  const int lane = threadIdx.x & 31;
  const _Float16* r = p + (lane & 15) * ld + ((lane >> 4) << 3);
  v16h a;
#pragma unroll
  for (int e = 0; e < 8; ++e) a[e] = r[e];
#pragma unroll
  for (int e = 0; e < 8; ++e) a[8 + e] = r[16 + e];
  return a;
}

// B-matrix fragment (32x16 f16) with B[k][n] = src[n*ld + k]  (source row n is
// output column n). half = lane>>4: e<8 -> k = half*8+e ; e>=8 -> k = 16 + half*8 + (e-8).
// Per-lane access is two runs of 8 contiguous halves (16B each).
__device__ __forceinline__ v16h frag_bt_ld(const _Float16* p, size_t ld) {
  const int lane = threadIdx.x & 31;
  const _Float16* r = p + (size_t)(lane & 15) * ld + ((lane >> 4) << 3);
  v16h b;
#pragma unroll
  for (int e = 0; e < 8; ++e) b[e] = r[e];
#pragma unroll
  for (int e = 0; e < 8; ++e) b[8 + e] = r[16 + e];
  return b;
}

// ---------------------------------------------------------------------------
__global__ void cvt_f32_to_f16(const float* __restrict__ src,
                               _Float16* __restrict__ dst, int n) {
  int i = blockIdx.x * blockDim.x + threadIdx.x;
  if (i < n) dst[i] = (_Float16)src[i];
}

// ---------------------------------------------------------------------------
// GEMM: acc = X[m,:] . W[n,:] (+bias)*scale, X:(B*L,D) f16, W:(D,D) f16.
// TR==0: Y[((b*H + n/64)*L + l)*64 + n%64]      (head-major rows)
// TR==1: Y[((b*H + n/64)*64 + n%64)*L + l]      (head-major, dh-major = vT)
template <int TR>
__global__ __launch_bounds__(128) void proj_kernel(
    const _Float16* __restrict__ X, const _Float16* __restrict__ W,
    const float* __restrict__ bias, _Float16* __restrict__ Y, float scale) {
  const int wv = threadIdx.x >> 5;
  const int m0 = blockIdx.x * 16;
  const int n0 = blockIdx.y * 64 + wv * 16;
  v8f acc = {};
  for (int kc = 0; kc < D_ / 32; ++kc) {
    v16h a = frag_a_ld(X + (size_t)m0 * D_ + kc * 32, D_);
    v16h b = frag_bt_ld(W + (size_t)n0 * D_ + kc * 32, D_);
    acc = wmma_f16(a, b, acc);
  }
  const int lane = threadIdx.x & 31;
  const int col  = lane & 15;
  const int gn   = n0 + col;
  const int h = gn >> 6, dh = gn & 63;
  const float bv = bias[gn];
#pragma unroll
  for (int e = 0; e < 8; ++e) {
    const int row = ((lane >> 4) << 3) + e;
    const int gm  = m0 + row;
    const int bb = gm >> 11, l = gm & (L_ - 1);     // L = 2048
    const _Float16 val = (_Float16)((acc[e] + bv) * scale);
    if (TR)
      Y[((size_t)(bb * H_ + h) * DH_ + dh) * L_ + l] = val;
    else
      Y[((size_t)(bb * H_ + h) * L_ + l) * DH_ + dh] = val;
  }
}

// ---------------------------------------------------------------------------
// out[gm, gn] = O_heads[gm, :] . fc_w[gn, :] + fc_b[gn]   (f32 out)
// O stored head-major: O[((b*H + k/64)*L + l)*64 + k%64], k = h*64+dh
__global__ __launch_bounds__(128) void fc_kernel(
    const _Float16* __restrict__ O, const _Float16* __restrict__ W,
    const float* __restrict__ bias, float* __restrict__ out) {
  const int wv = threadIdx.x >> 5;
  const int m0 = blockIdx.x * 16;
  const int n0 = blockIdx.y * 64 + wv * 16;
  const int lane = threadIdx.x & 31;
  const int arow = lane & 15;
  const int gm_a = m0 + arow;
  const int bb = gm_a >> 11, l = gm_a & (L_ - 1);
  const _Float16* obase = O + ((size_t)bb * H_ * L_ + l) * (size_t)DH_;
  v8f acc = {};
  for (int kc = 0; kc < D_ / 32; ++kc) {
    const int c0 = kc * 32 + ((lane >> 4) << 3);
    const int c1 = c0 + 16;
    const _Float16* p0 = obase + (size_t)(c0 >> 6) * (L_ * DH_) + (c0 & 63);
    const _Float16* p1 = obase + (size_t)(c1 >> 6) * (L_ * DH_) + (c1 & 63);
    v16h a;
#pragma unroll
    for (int e = 0; e < 8; ++e) a[e] = p0[e];
#pragma unroll
    for (int e = 0; e < 8; ++e) a[8 + e] = p1[e];
    v16h b = frag_bt_ld(W + (size_t)n0 * D_ + kc * 32, D_);
    acc = wmma_f16(a, b, acc);
  }
  const int col = lane & 15;
  const int gn  = n0 + col;
  const float bv = bias[gn];
#pragma unroll
  for (int e = 0; e < 8; ++e) {
    const int gm = m0 + ((lane >> 4) << 3) + e;
    out[(size_t)gm * D_ + gn] = acc[e] + bv;
  }
}

// ---------------------------------------------------------------------------
// One workgroup per (b, h, 16-row query block). 8 waves.
// Srel[i,j] = q_i . E[(L-1) - i + j]   (reference skew trick, closed form)
__global__ __launch_bounds__(256) void attn_kernel(
    const _Float16* __restrict__ qh, const _Float16* __restrict__ kh,
    const _Float16* __restrict__ vT, const _Float16* __restrict__ E16,
    float* __restrict__ attn_out, _Float16* __restrict__ o_tmp) {
  extern __shared__ float smem[];
  float* sc     = smem;                     // 16 x SSTRIDE logits / attn panel
  float* spanel = sc + 16 * SSTRIDE;        // per-wave 16x32 Q.E^T panel (+pad)
  float* sred   = spanel + 8 * 16 * 33;     // 4 x 16 x 16 split-K reduction

  const int ib = blockIdx.x;                // query 16-row block index
  const int bh = blockIdx.z * H_ + blockIdx.y;
  const int i0 = ib * 16;
  const int wv = threadIdx.x >> 5;
  const int lane = threadIdx.x & 31;
  const int col = lane & 15;
  const int rbase = (lane >> 4) << 3;

  const _Float16* qrow = qh + ((size_t)bh * L_ + i0) * DH_;
  const v16h aq0 = frag_a_ld(qrow, DH_);
  const v16h aq1 = frag_a_ld(qrow + 32, DH_);
  float* mypan = spanel + wv * (16 * 33);

  // ---- phase 1: scores = QK^T + Srel (pre-scaled by 1/sqrt(DH) via qh) ----
  for (int jt = wv; jt <= ib; jt += 8) {
    const _Float16* krow = kh + ((size_t)bh * L_ + jt * 16) * DH_;
    if (jt + 8 <= ib)
      __builtin_prefetch(kh + ((size_t)bh * L_ + (jt + 8) * 16) * DH_, 0, 1);
    const int base = L_ - 16 - 16 * (ib - jt);      // >= 0 for jt <= ib
    const _Float16* erow = E16 + (size_t)base * DH_;

    v8f s = {}, p0 = {}, p1 = {};
    v16h bk = frag_bt_ld(krow, DH_);
    s = wmma_f16(aq0, bk, s);
    v16h be0 = frag_bt_ld(erow, DH_);
    p0 = wmma_f16(aq0, be0, p0);
    v16h be1 = frag_bt_ld(erow + 16 * DH_, DH_);
    p1 = wmma_f16(aq0, be1, p1);
    bk = frag_bt_ld(krow + 32, DH_);
    s = wmma_f16(aq1, bk, s);
    be0 = frag_bt_ld(erow + 32, DH_);
    p0 = wmma_f16(aq1, be0, p0);
    be1 = frag_bt_ld(erow + 16 * DH_ + 32, DH_);
    p1 = wmma_f16(aq1, be1, p1);

#pragma unroll
    for (int e = 0; e < 8; ++e) {
      mypan[(rbase + e) * 33 + col]      = p0[e];
      mypan[(rbase + e) * 33 + 16 + col] = p1[e];
    }
    // same-wave LDS RAW: DS ops are in order within a wave (ISA 7.3)
#pragma unroll
    for (int e = 0; e < 8; ++e) {
      const int row = rbase + e;
      float f = s[e] + mypan[row * 33 + (col - row + 15)];   // Srel shift
      if (jt == ib && col > row) f = -1.0e9f;                // causal mask
      sc[row * SSTRIDE + jt * 16 + col] = f;
    }
  }
  __syncthreads();

  // ---- phase 2: row softmax; write attn to HBM exactly once ----
  const int ncols = (ib + 1) * 16;
  const int pad32 = (ncols + 31) & ~31;
  {
    const int r   = threadIdx.x >> 4;
    const int sub = threadIdx.x & 15;
    float* srow = sc + r * SSTRIDE;
    float mx = -3.0e38f;
    for (int c = sub; c < ncols; c += 16) mx = fmaxf(mx, srow[c]);
#pragma unroll
    for (int off = 8; off > 0; off >>= 1) mx = fmaxf(mx, __shfl_xor(mx, off, 16));
    float sum = 0.f;
    for (int c = sub; c < ncols; c += 16) {
      float ev = __expf(srow[c] - mx);
      srow[c] = ev;
      sum += ev;
    }
#pragma unroll
    for (int off = 8; off > 0; off >>= 1) sum += __shfl_xor(sum, off, 16);
    const float inv = 1.0f / sum;
    float* ag = attn_out + ((size_t)bh * L_ + (i0 + r)) * (size_t)L_;
    for (int c = sub; c < ncols; c += 16) {
      float a = srow[c] * inv;
      srow[c] = a;
      ag[c] = a;
    }
    for (int c = ncols + sub; c < pad32; c += 16) srow[c] = 0.f;  // K pad
    for (int c = ncols + sub; c < L_; c += 16) ag[c] = 0.f;       // masked tail
  }
  __syncthreads();

  // ---- phase 3: O = attn x V (B from vT: two contiguous 16B runs per lane) ----
  {
    const int nsub = wv & 3;          // 16-col chunk of DH
    const int kpar = wv >> 2;         // split-K half
    const int nk32 = pad32 >> 5;
    const _Float16* vtile =
        vT + ((size_t)bh * DH_ + nsub * 16) * (size_t)L_;
    v8f acc = {};
    for (int kc = kpar; kc < nk32; kc += 2) {
      v16h a;
#pragma unroll
      for (int e = 0; e < 8; ++e)
        a[e] = (_Float16)sc[(lane & 15) * SSTRIDE + kc * 32 + rbase + e];
#pragma unroll
      for (int e = 0; e < 8; ++e)
        a[8 + e] = (_Float16)sc[(lane & 15) * SSTRIDE + kc * 32 + 16 + rbase + e];
      // B[k][n] = V[kc*32+k][nsub*16+n] = vT[(bh*DH + nsub*16 + n)*L + kc*32+k]
      v16h bv = frag_bt_ld(vtile + kc * 32, (size_t)L_);
      acc = wmma_f16(a, bv, acc);
    }
    if (kpar == 1) {
#pragma unroll
      for (int e = 0; e < 8; ++e)
        sred[nsub * 256 + (rbase + e) * 16 + col] = acc[e];
    }
    __syncthreads();
    if (kpar == 0) {
      _Float16* orow = o_tmp + ((size_t)bh * L_ + i0) * DH_ + nsub * 16;
#pragma unroll
      for (int e = 0; e < 8; ++e) {
        float v = acc[e] + sred[nsub * 256 + (rbase + e) * 16 + col];
        orow[(size_t)(rbase + e) * DH_ + col] = (_Float16)v;
      }
    }
  }
}

// ---------------------------------------------------------------------------
extern "C" void kernel_launch(void* const* d_in, const int* in_sizes, int n_in,
                              void* d_out, int out_size, void* d_ws,
                              size_t ws_size, hipStream_t stream) {
  (void)in_sizes; (void)n_in; (void)out_size; (void)ws_size;

  const float* q    = (const float*)d_in[0];
  const float* k    = (const float*)d_in[1];
  const float* v    = (const float*)d_in[2];
  const float* Wq_w = (const float*)d_in[3];
  const float* Wq_b = (const float*)d_in[4];
  const float* Wk_w = (const float*)d_in[5];
  const float* Wk_b = (const float*)d_in[6];
  const float* Wv_w = (const float*)d_in[7];
  const float* Wv_b = (const float*)d_in[8];
  const float* fc_w = (const float*)d_in[9];
  const float* fc_b = (const float*)d_in[10];
  const float* E    = (const float*)d_in[11];

  const int NQ = B_ * L_ * D_;        // 2,097,152
  const int NW = D_ * D_;             //   262,144
  const int NE = L_ * DH_;            //   131,072
  const int NH = B_ * H_ * L_ * DH_;  // 2,097,152

  _Float16* w    = (_Float16*)d_ws;
  _Float16* q16  = w;            w += NQ;
  _Float16* k16  = w;            w += NQ;
  _Float16* v16  = w;            w += NQ;
  _Float16* wq16 = w;            w += NW;
  _Float16* wk16 = w;            w += NW;
  _Float16* wv16 = w;            w += NW;
  _Float16* fw16 = w;            w += NW;
  _Float16* e16  = w;            w += NE;
  _Float16* qhp  = w;            w += NH;
  _Float16* khp  = w;            w += NH;
  _Float16* vtp  = w;            w += NH;   // V transposed per head
  _Float16* otmp = w;            w += NH;

  auto cvt = [&](const float* s, _Float16* d, int n) {
    cvt_f32_to_f16<<<(n + 255) / 256, 256, 0, stream>>>(s, d, n);
  };
  cvt(q, q16, NQ);  cvt(k, k16, NQ);  cvt(v, v16, NQ);
  cvt(Wq_w, wq16, NW); cvt(Wk_w, wk16, NW); cvt(Wv_w, wv16, NW);
  cvt(fc_w, fw16, NW); cvt(E, e16, NE);

  dim3 pg(B_ * L_ / 16, D_ / 64), pb(128);
  proj_kernel<0><<<pg, pb, 0, stream>>>(q16, wq16, Wq_b, qhp, 0.125f);
  proj_kernel<0><<<pg, pb, 0, stream>>>(k16, wk16, Wk_b, khp, 1.0f);
  proj_kernel<1><<<pg, pb, 0, stream>>>(v16, wv16, Wv_b, vtp, 1.0f);

  float* attn_out = (float*)d_out + NQ;
  const size_t lds_bytes =
      (size_t)(16 * SSTRIDE + 8 * 16 * 33 + 4 * 256) * sizeof(float); // ~150 KB
  dim3 ag(L_ / 16, H_, B_), ab(256);
  attn_kernel<<<ag, ab, lds_bytes, stream>>>(qhp, khp, vtp, e16, attn_out, otmp);

  fc_kernel<<<pg, pb, 0, stream>>>(otmp, fw16, fc_b, (float*)d_out);
}
